// WGcn_73632919322687
// MI455X (gfx1250) — matrix operand order repaired and checked by reading, not compiled, not run on hardware
//
#include <hip/hip_runtime.h>
#include <hip/hip_bf16.h>
#include <math.h>

// ---------------------------------------------------------------------------
// HeterSumGraph-style 3-layer GAT for MI455X (gfx1250, wave32, WMMA).
// Dense GEMMs run on v_wmma_f32_16x16x32_f16 with f32 accumulation.
// LDS tiles are stored PRE-SWIZZLED into WMMA fragment layout so each lane's
// fragment is a contiguous, 32B-aligned v16h -> 2x ds_load_b128 per operand.
// Row-guarded tile loads are hoisted-branch + b128 vector loads (no per-elem
// exec masking). FFN is fully fused (relu-mid tile never leaves LDS).
// Edge softmax: encoded-uint atomicMax + f32 global atomic adds.
// ---------------------------------------------------------------------------

typedef _Float16 f16;
typedef __attribute__((ext_vector_type(4)))  float    v4f;
typedef __attribute__((ext_vector_type(8)))  _Float16 v8h;
typedef __attribute__((ext_vector_type(16))) _Float16 v16h;
typedef __attribute__((ext_vector_type(8)))  float    v8f;

#define NW   50000
#define NS   100000
#define NN   2000
#define DIM  256
#define HID  64
#define FE   64
#define NHEAD 8
#define FFN  512
#define EWW  500000
#define EWS  500000
#define ESN  100000

static inline unsigned cdiv(size_t a, size_t b) { return (unsigned)((a + b - 1) / b); }

// Fragment layouts (16-bit WMMA operands per CDNA5 ISA):
// A 16x32: lane = ((k>>3)&1)*16 + (m&15), slot = (k&7) + ((k&16)?8:0)
// B 32x16: lane = ((k>>3)&1)*16 + (n&15), slot = (k&7) + ((k&16)?8:0)
// An 8-wide aligned K chunk maps to 8 contiguous slots of one lane.

// convert 8 contiguous f32 (16B-aligned) -> v8h, with hoisted row guard
__device__ inline v8h cvt8(const float* gp, bool ok) {
    v8h h = {};
    if (ok) {
        v4f f0 = ((const v4f*)gp)[0];
        v4f f1 = ((const v4f*)gp)[1];
#pragma unroll
        for (int j = 0; j < 4; ++j) { h[j] = (f16)f0[j]; h[4 + j] = (f16)f1[j]; }
    }
    return h;
}

// ---------------- generic WMMA GEMM:  C = act(A@B + bias) ------------------
// A: [M,K] f32 (row tail guarded), B: [K,N] f32, K%32==0, N%64==0.
// 256 thr = 8 waves; block tile 64x64; wave -> 16x32 (2 accumulators).
__global__ __launch_bounds__(256) void gemm_wmma_kernel(
    const float* __restrict__ A, const float* __restrict__ B,
    const float* __restrict__ bias, float* __restrict__ C,
    int M, int N, int K, int act)
{
    __shared__ __align__(32) f16 As[4 * 32 * 16];  // 4 row-blocks of frags
    __shared__ __align__(32) f16 Bs[4 * 32 * 16];  // 4 col-blocks of frags
    int tid = threadIdx.x, wave = tid >> 5, lane = tid & 31;
    int m0 = blockIdx.y * 64, n0 = blockIdx.x * 64;
    int mo = (wave >> 1) * 16, no = (wave & 1) * 32;
    int rb = mo >> 4;
    v8f acc0 = {}; v8f acc1 = {};

    // per-thread staging units (exactly 256 each)
    int am = tid >> 2, akc = (tid & 3) * 8;          // A: row m, k-chunk
    int bn = tid & 63, bkc = (tid >> 6) * 8;         // B: col n, k-chunk
    int a_kh = (akc >> 3) & 1, a_jb = (akc & 16) ? 8 : 0;
    f16* a_dst = &As[(((am >> 4) * 32) + (a_kh << 4) + (am & 15)) * 16 + a_jb];
    f16* b_dst = &Bs[(((bn >> 4) * 32) + (bn & 15) + (((bkc >> 3) & 1) << 4)) * 16 + ((bkc & 16) ? 8 : 0)];
    int gr = m0 + am;
    bool arow_ok = gr < M;
    const float* a_src = A + (size_t)gr * K + akc;

    for (int k0 = 0; k0 < K; k0 += 32) {
        *(v8h*)a_dst = cvt8(a_src + k0, arow_ok);
        {
            const float* gp = B + (size_t)(k0 + bkc) * N + n0 + bn;
            v8h h;
#pragma unroll
            for (int j = 0; j < 8; ++j) h[j] = (f16)gp[(size_t)j * N];
            *(v8h*)b_dst = h;
        }
        __syncthreads();
        v16h a  = ((const v16h*)As)[rb * 32 + lane];
        v16h b0 = ((const v16h*)Bs)[(no >> 4) * 32 + lane];
        v16h b1 = ((const v16h*)Bs)[((no >> 4) + 1) * 32 + lane];
        acc0 = __builtin_amdgcn_wmma_f32_16x16x32_f16(false, a, false, b0, (short)0, acc0, false, false);
        acc1 = __builtin_amdgcn_wmma_f32_16x16x32_f16(false, a, false, b1, (short)0, acc1, false, false);
        __syncthreads();
    }
    int r = lane & 15, kh = lane >> 4;
#pragma unroll
    for (int i = 0; i < 8; ++i) {
        int row = m0 + mo + i + kh * 8;
        if (row < M) {
            int c0 = n0 + no + r, c1 = c0 + 16;
            float v0 = acc0[i] + (bias ? bias[c0] : 0.0f);
            float v1 = acc1[i] + (bias ? bias[c1] : 0.0f);
            if (act == 1) { v0 = fmaxf(v0, 0.0f); v1 = fmaxf(v1, 0.0f); }
            C[(size_t)row * N + c0] = v0;
            C[(size_t)row * N + c1] = v1;
        }
    }
}

// ---------------- fused FFN: Y = relu(H@w1+b1)@w2 + b2 + H (pre-LN) --------
// dw in {64,256}, FFN=512. 32-row tile. All LDS tiles fragment-swizzled.
// Phase 1: 128-col chunks, wave = 16x32 (2 wmma / barrier pair).
// Phase 2: 64-col chunks, wave = 16x16.
__global__ __launch_bounds__(256) void ffn_fused_kernel(
    const float* __restrict__ H, const float* __restrict__ w1,
    const float* __restrict__ b1, const float* __restrict__ w2,
    const float* __restrict__ b2, float* __restrict__ Y, int M, int dw)
{
    __shared__ __align__(32) f16 hA[8 * 2 * 32 * 16];   // K<=256: [kb][rb][frag] 16KB
    __shared__ __align__(32) f16 mid[16 * 2 * 32 * 16]; // K=512:  [kb][rb][frag] 32KB
    __shared__ __align__(32) f16 Bs[8 * 32 * 16];       // up to 32x128 weight tile 8KB
    int tid = threadIdx.x, wave = tid >> 5, lane = tid & 31;
    int m0 = blockIdx.x * 32;
    int mo = (wave >> 2) * 16;
    int rb = mo >> 4;
    int r = lane & 15, kh = lane >> 4;

    // stage H tile (swizzled): 32 rows x dw cols, 8-wide K chunks
    {
        int chunks = dw >> 3;
        for (int u = tid; u < 32 * chunks; u += 256) {
            int m = u / chunks, kc = (u % chunks) * 8;
            int kb = kc >> 5, kk = kc & 31;
            int khh = (kk >> 3) & 1, jb = (kk & 16) ? 8 : 0;
            int gr = m0 + m;
            *(v8h*)&hA[(((kb * 2 + (m >> 4)) * 32) + (khh << 4) + (m & 15)) * 16 + jb] =
                cvt8(H + (size_t)gr * dw + kc, gr < M);
        }
    }
    __syncthreads();

    // phase 1: mid = relu(H@w1 + b1)  [32,512]; 128-col chunks, 2 accs/wave
    {
        int no = (wave & 3) * 32;
        for (int n1 = 0; n1 < FFN; n1 += 128) {
            v8f acc0 = {}; v8f acc1 = {};
            for (int k0 = 0; k0 < dw; k0 += 32) {
                // stage 32x128 tile of w1 (512 units, 2 per thread)
#pragma unroll
                for (int uu = 0; uu < 2; ++uu) {
                    int u = tid + uu * 256;
                    int n = u & 127, kc = (u >> 7) * 8;
                    const float* gp = w1 + (size_t)(k0 + kc) * FFN + n1 + n;
                    v8h h;
#pragma unroll
                    for (int j = 0; j < 8; ++j) h[j] = (f16)gp[(size_t)j * FFN];
                    *(v8h*)&Bs[(((n >> 4) * 32) + (n & 15) + (((kc >> 3) & 1) << 4)) * 16 + ((kc & 16) ? 8 : 0)] = h;
                }
                __syncthreads();
                v16h a  = ((const v16h*)hA)[((k0 >> 5) * 2 + rb) * 32 + lane];
                v16h b0 = ((const v16h*)Bs)[(no >> 4) * 32 + lane];
                v16h b1 = ((const v16h*)Bs)[((no >> 4) + 1) * 32 + lane];
                acc0 = __builtin_amdgcn_wmma_f32_16x16x32_f16(false, a, false, b0, (short)0, acc0, false, false);
                acc1 = __builtin_amdgcn_wmma_f32_16x16x32_f16(false, a, false, b1, (short)0, acc1, false, false);
                __syncthreads();
            }
            // store both 16x16 results into mid, pre-swizzled as phase-2 A.
#pragma unroll
            for (int half = 0; half < 2; ++half) {
                int kcol = n1 + no + half * 16 + r;     // phase-2 K index
                int kb = kcol >> 5, kk = kcol & 31;
                int khd = (kk >> 3) & 1;
                int j = (kk & 7) + ((kk & 16) ? 8 : 0);
                f16* mp = &mid[(((kb * 2 + rb) * 32) + (khd << 4)) * 16 + j];
                float bb = b1[kcol];
                const v8f& acc = half ? acc1 : acc0;
#pragma unroll
                for (int i = 0; i < 8; ++i)
                    mp[(size_t)(i + kh * 8) * 16] = (f16)fmaxf(acc[i] + bb, 0.0f);
            }
        }
    }
    __syncthreads();

    // phase 2: Y = mid@w2 + b2 + H ; 64-col chunks, 1 acc/wave
    {
        int no = (wave & 3) * 16;
        int bn = tid & 63, bkc = (tid >> 6) * 8;
        f16* b_dst = &Bs[(((bn >> 4) * 32) + (bn & 15) + (((bkc >> 3) & 1) << 4)) * 16 + ((bkc & 16) ? 8 : 0)];
        for (int n2 = 0; n2 < dw; n2 += 64) {
            v8f acc = {};
            for (int k0 = 0; k0 < FFN; k0 += 32) {
                {
                    const float* gp = w2 + (size_t)(k0 + bkc) * dw + n2 + bn;
                    v8h h;
#pragma unroll
                    for (int j = 0; j < 8; ++j) h[j] = (f16)gp[(size_t)j * dw];
                    *(v8h*)b_dst = h;
                }
                __syncthreads();
                v16h a = ((const v16h*)mid)[((k0 >> 5) * 2 + rb) * 32 + lane];
                v16h b = ((const v16h*)Bs)[(no >> 4) * 32 + lane];
                acc = __builtin_amdgcn_wmma_f32_16x16x32_f16(false, a, false, b, (short)0, acc, false, false);
                __syncthreads();
            }
#pragma unroll
            for (int i = 0; i < 8; ++i) {
                int row = m0 + mo + i + kh * 8;
                int col = n2 + no + r;
                if (row < M)
                    Y[(size_t)row * dw + col] = acc[i] + b2[col] + H[(size_t)row * dw + col];
            }
        }
    }
}

// ---------------- LayerNorm (wave-per-row, wave32 shfl reductions) ---------
__global__ __launch_bounds__(256) void layernorm_kernel(
    float* __restrict__ X, const float* __restrict__ g,
    const float* __restrict__ b, int M, int dw)
{
    int wave = threadIdx.x >> 5, lane = threadIdx.x & 31;
    int row = blockIdx.x * 8 + wave;
    if (row >= M) return;
    float v[8];
    int nv = dw / 32;
    float s = 0.0f;
    for (int i = 0; i < nv; ++i) { v[i] = X[(size_t)row * dw + i * 32 + lane]; s += v[i]; }
    for (int off = 16; off > 0; off >>= 1) s += __shfl_xor(s, off, 32);
    float mu = s / dw;
    float q = 0.0f;
    for (int i = 0; i < nv; ++i) { float d = v[i] - mu; q += d * d; }
    for (int off = 16; off > 0; off >>= 1) q += __shfl_xor(q, off, 32);
    float inv = rsqrtf(q / dw + 1e-5f);
    for (int i = 0; i < nv; ++i) {
        int c = i * 32 + lane;
        X[(size_t)row * dw + c] = (v[i] - mu) * inv * g[c] + b[c];
    }
}

// ---------------- small elementwise / gather kernels -----------------------
__global__ void gather_embed_kernel(const int* __restrict__ wid,
                                    const float* __restrict__ embed,
                                    float* __restrict__ wf)
{
    size_t i = (size_t)blockIdx.x * blockDim.x + threadIdx.x;
    if (i >= (size_t)NW * DIM) return;
    size_t n = i / DIM, c = i % DIM;
    wf[i] = embed[(size_t)wid[n] * DIM + c];
}

__global__ void fill_u32_kernel(unsigned* __restrict__ p, unsigned v, size_t n)
{
    size_t i = (size_t)blockIdx.x * blockDim.x + threadIdx.x;
    if (i < n) p[i] = v;
}

__global__ void elu_add_kernel(const float* __restrict__ A,
                               const float* __restrict__ origin,
                               float* __restrict__ O, size_t n)
{
    size_t i = (size_t)blockIdx.x * blockDim.x + threadIdx.x;
    if (i >= n) return;
    float x = A[i];
    float e = (x > 0.0f) ? x : (expf(x) - 1.0f);
    O[i] = e + (origin ? origin[i] : 0.0f);
}

// ez[n,h] = sum_k z[n,h,k] * attnZ[h,k]
__global__ void ez_kernel(const float* __restrict__ z, const float* __restrict__ attnZ,
                          float* __restrict__ ez, int N, int dh)
{
    int i = blockIdx.x * blockDim.x + threadIdx.x;
    if (i >= N * NHEAD) return;
    int n = i >> 3, h = i & 7;
    const float* zp = z + (size_t)n * (NHEAD * dh) + h * dh;
    const float* ap = attnZ + h * dh;
    float s = 0.0f;
    for (int k = 0; k < dh; ++k) s += zp[k] * ap[k];
    ez[i] = s;
}

// ef[t,h] = sum_k (tfembed[t] @ featW[:,h,k] + featB[h,k]) * attnF[h,k], t<10
__global__ void ef_kernel(const float* __restrict__ tfembed, const float* __restrict__ featW,
                          const float* __restrict__ featB, const float* __restrict__ attnF,
                          float* __restrict__ ef, int dh)
{
    int i = threadIdx.x;
    if (i >= 10 * NHEAD) return;
    int t = i >> 3, h = i & 7;
    float s = 0.0f;
    for (int k = 0; k < dh; ++k) {
        float acc = featB[h * dh + k];
        for (int d = 0; d < FE; ++d)
            acc += tfembed[t * FE + d] * featW[(size_t)d * (NHEAD * dh) + h * dh + k];
        s += acc * attnF[h * dh + k];
    }
    ef[i] = s;
}

// order-preserving float<->uint encode for atomicMax on floats
__device__ inline unsigned encf(float f) {
    unsigned u = __float_as_uint(f);
    return (u & 0x80000000u) ? ~u : (u | 0x80000000u);
}
__device__ inline float decf(unsigned u) {
    return (u & 0x80000000u) ? __uint_as_float(u & 0x7fffffffu) : __uint_as_float(~u);
}

__global__ void edge_max_kernel(const int* __restrict__ src, const int* __restrict__ dst,
                                const int* __restrict__ tff, const float* __restrict__ ez,
                                const float* __restrict__ ef, unsigned* __restrict__ menc, int E)
{
    int i = blockIdx.x * blockDim.x + threadIdx.x;
    if (i >= E * NHEAD) return;
    int e = i >> 3, h = i & 7;
    float v = ez[src[e] * NHEAD + h] + ef[tff[e] * NHEAD + h];
    v = (v > 0.0f) ? v : 0.01f * v;            // leaky_relu(0.01)
    atomicMax(&menc[dst[e] * NHEAD + h], encf(v));
}

__global__ void edge_exp_kernel(const int* __restrict__ src, const int* __restrict__ dst,
                                const int* __restrict__ tff, const float* __restrict__ ez,
                                const float* __restrict__ ef, const unsigned* __restrict__ menc,
                                float* __restrict__ ex, float* __restrict__ den, int E)
{
    int i = blockIdx.x * blockDim.x + threadIdx.x;
    if (i >= E * NHEAD) return;
    int e = i >> 3, h = i & 7;
    float v = ez[src[e] * NHEAD + h] + ef[tff[e] * NHEAD + h];
    v = (v > 0.0f) ? v : 0.01f * v;
    float m = decf(menc[dst[e] * NHEAD + h]);
    float x = expf(v - m);
    ex[i] = x;
    atomicAdd(&den[dst[e] * NHEAD + h], x);
}

__global__ void edge_agg_kernel(const int* __restrict__ src, const int* __restrict__ dst,
                                const float* __restrict__ ex, const float* __restrict__ den,
                                const float* __restrict__ z, float* __restrict__ out,
                                int E, int dh)
{
    int i = blockIdx.x * blockDim.x + threadIdx.x;
    if (i >= E * NHEAD) return;
    int e = i >> 3, h = i & 7;
    int d = dst[e];
    float alpha = ex[i] / den[d * NHEAD + h];
    const float* zp = z + (size_t)src[e] * (NHEAD * dh) + h * dh;
    float* op = out + (size_t)d * (NHEAD * dh) + h * dh;
    for (int k = 0; k < dh; ++k) atomicAdd(&op[k], alpha * zp[k]);
}

__global__ void final_head_kernel(const float* __restrict__ news, const float* __restrict__ w,
                                  const float* __restrict__ b, float* __restrict__ out)
{
    int i = blockIdx.x * blockDim.x + threadIdx.x;
    if (i >= NN) return;
    float s = b[0];
    for (int k = 0; k < FE; ++k) s += news[(size_t)i * FE + k] * w[k];
    out[i] = s;
}

// ---------------------------------------------------------------------------
extern "C" void kernel_launch(void* const* d_in, const int* in_sizes, int n_in,
                              void* d_out, int out_size, void* d_ws, size_t ws_size,
                              hipStream_t stream)
{
    (void)in_sizes; (void)n_in; (void)out_size; (void)ws_size;
    const int* wid       = (const int*)d_in[0];
    const int* ww_src    = (const int*)d_in[1];
    const int* ww_dst    = (const int*)d_in[2];
    const int* tffrac_ww = (const int*)d_in[3];
    const int* ws_src    = (const int*)d_in[4];
    const int* ws_dst    = (const int*)d_in[5];
    const int* tffrac_ws = (const int*)d_in[6];
    const int* sn_src    = (const int*)d_in[7];
    const int* sn_dst    = (const int*)d_in[8];
    const int* tffrac_sn = (const int*)d_in[9];
    const float* embed   = (const float*)d_in[10];
    const float* tfembed = (const float*)d_in[11];
    // layers: fcW featW featB attnZ attnF w1 b1 w2 b2 g b  (11 each)
    auto LP = [&](int layer, int j) -> const float* { return (const float*)d_in[12 + layer * 11 + j]; };
    const float* wh_w = (const float*)d_in[45];
    const float* wh_b = (const float*)d_in[46];

    // ----- workspace carve-out (floats); peak ~347 MB -----
    float* wsf = (float*)d_ws;
    size_t off = 0;
    auto alloc = [&](size_t n) { float* p = wsf + off; off += n; return p; };
    float* wf   = alloc((size_t)NW * DIM);
    float* z1   = alloc((size_t)NW * DIM);
    float* agg1 = alloc((size_t)NW * DIM);
    float* y1   = alloc((size_t)NW * DIM);
    float* ez1  = alloc((size_t)NW * NHEAD);
    unsigned* m1 = (unsigned*)alloc((size_t)NW * NHEAD);
    float* den1 = alloc((size_t)NW * NHEAD);
    float* ex1  = alloc((size_t)EWW * NHEAD);
    float* ef   = alloc(128);
    float* z2   = alloc((size_t)NW * HID);
    float* ez2  = alloc((size_t)NW * NHEAD);
    unsigned* m2 = (unsigned*)alloc((size_t)NS * NHEAD);
    float* den2 = alloc((size_t)NS * NHEAD);
    float* ex2  = alloc((size_t)EWS * NHEAD);
    float* agg2 = alloc((size_t)NS * HID);
    float* y2   = alloc((size_t)NS * HID);
    float* z3   = alloc((size_t)NS * FE);
    float* ez3  = alloc((size_t)NS * NHEAD);
    unsigned* m3 = (unsigned*)alloc((size_t)NN * NHEAD);
    float* den3 = alloc((size_t)NN * NHEAD);
    float* ex3  = alloc((size_t)ESN * NHEAD);
    float* agg3 = alloc((size_t)NN * FE);
    float* y3   = alloc((size_t)NN * FE);

    const int T = 256;
    dim3 blk(T);

    // ================= set_wnfeature =================
    gather_embed_kernel<<<cdiv((size_t)NW * DIM, T), blk, 0, stream>>>(wid, embed, wf);

    // ================= layer 1: word2word (dout=256, dh=32) =================
    {
        const int dh = 32, dw = DIM, M = NW, E = EWW, ND = NW;
        gemm_wmma_kernel<<<dim3(dw / 64, cdiv(M, 64)), blk, 0, stream>>>(
            wf, LP(0, 0), nullptr, z1, M, dw, DIM, 0);
        ez_kernel<<<cdiv((size_t)M * NHEAD, T), blk, 0, stream>>>(z1, LP(0, 3), ez1, M, dh);
        ef_kernel<<<1, 128, 0, stream>>>(tfembed, LP(0, 1), LP(0, 2), LP(0, 4), ef, dh);
        fill_u32_kernel<<<cdiv((size_t)ND * NHEAD, T), blk, 0, stream>>>(m1, 0u, (size_t)ND * NHEAD);
        fill_u32_kernel<<<cdiv((size_t)ND * NHEAD, T), blk, 0, stream>>>((unsigned*)den1, 0u, (size_t)ND * NHEAD);
        edge_max_kernel<<<cdiv((size_t)E * NHEAD, T), blk, 0, stream>>>(ww_src, ww_dst, tffrac_ww, ez1, ef, m1, E);
        edge_exp_kernel<<<cdiv((size_t)E * NHEAD, T), blk, 0, stream>>>(ww_src, ww_dst, tffrac_ww, ez1, ef, m1, ex1, den1, E);
        fill_u32_kernel<<<cdiv((size_t)ND * dw, T), blk, 0, stream>>>((unsigned*)agg1, 0u, (size_t)ND * dw);
        edge_agg_kernel<<<cdiv((size_t)E * NHEAD, T), blk, 0, stream>>>(ww_src, ww_dst, ex1, den1, z1, agg1, E, dh);
        elu_add_kernel<<<cdiv((size_t)ND * dw, T), blk, 0, stream>>>(agg1, wf, agg1, (size_t)ND * dw);
        ffn_fused_kernel<<<cdiv(ND, 32), blk, 0, stream>>>(agg1, LP(0, 5), LP(0, 6), LP(0, 7), LP(0, 8), y1, ND, dw);
        layernorm_kernel<<<cdiv(ND, 8), blk, 0, stream>>>(y1, LP(0, 9), LP(0, 10), ND, dw);
    }

    // ================= layer 2: word2sent (dout=64, dh=8) =================
    {
        const int dh = 8, dw = HID, M = NW, E = EWS, ND = NS;
        gemm_wmma_kernel<<<dim3(dw / 64, cdiv(M, 64)), blk, 0, stream>>>(
            y1, LP(1, 0), nullptr, z2, M, dw, DIM, 0);
        ez_kernel<<<cdiv((size_t)M * NHEAD, T), blk, 0, stream>>>(z2, LP(1, 3), ez2, M, dh);
        ef_kernel<<<1, 128, 0, stream>>>(tfembed, LP(1, 1), LP(1, 2), LP(1, 4), ef, dh);
        fill_u32_kernel<<<cdiv((size_t)ND * NHEAD, T), blk, 0, stream>>>(m2, 0u, (size_t)ND * NHEAD);
        fill_u32_kernel<<<cdiv((size_t)ND * NHEAD, T), blk, 0, stream>>>((unsigned*)den2, 0u, (size_t)ND * NHEAD);
        edge_max_kernel<<<cdiv((size_t)E * NHEAD, T), blk, 0, stream>>>(ws_src, ws_dst, tffrac_ws, ez2, ef, m2, E);
        edge_exp_kernel<<<cdiv((size_t)E * NHEAD, T), blk, 0, stream>>>(ws_src, ws_dst, tffrac_ws, ez2, ef, m2, ex2, den2, E);
        fill_u32_kernel<<<cdiv((size_t)ND * dw, T), blk, 0, stream>>>((unsigned*)agg2, 0u, (size_t)ND * dw);
        edge_agg_kernel<<<cdiv((size_t)E * NHEAD, T), blk, 0, stream>>>(ws_src, ws_dst, ex2, den2, z2, agg2, E, dh);
        elu_add_kernel<<<cdiv((size_t)ND * dw, T), blk, 0, stream>>>(agg2, nullptr, agg2, (size_t)ND * dw);
        ffn_fused_kernel<<<cdiv(ND, 32), blk, 0, stream>>>(agg2, LP(1, 5), LP(1, 6), LP(1, 7), LP(1, 8), y2, ND, dw);
        layernorm_kernel<<<cdiv(ND, 8), blk, 0, stream>>>(y2, LP(1, 9), LP(1, 10), ND, dw);
    }

    // ================= layer 3: sent2news (dout=64, dh=8) =================
    {
        const int dh = 8, dw = FE, M = NS, E = ESN, ND = NN;
        gemm_wmma_kernel<<<dim3(dw / 64, cdiv(M, 64)), blk, 0, stream>>>(
            y2, LP(2, 0), nullptr, z3, M, dw, HID, 0);
        ez_kernel<<<cdiv((size_t)M * NHEAD, T), blk, 0, stream>>>(z3, LP(2, 3), ez3, M, dh);
        ef_kernel<<<1, 128, 0, stream>>>(tfembed, LP(2, 1), LP(2, 2), LP(2, 4), ef, dh);
        fill_u32_kernel<<<cdiv((size_t)ND * NHEAD, T), blk, 0, stream>>>(m3, 0u, (size_t)ND * NHEAD);
        fill_u32_kernel<<<cdiv((size_t)ND * NHEAD, T), blk, 0, stream>>>((unsigned*)den3, 0u, (size_t)ND * NHEAD);
        edge_max_kernel<<<cdiv((size_t)E * NHEAD, T), blk, 0, stream>>>(sn_src, sn_dst, tffrac_sn, ez3, ef, m3, E);
        edge_exp_kernel<<<cdiv((size_t)E * NHEAD, T), blk, 0, stream>>>(sn_src, sn_dst, tffrac_sn, ez3, ef, m3, ex3, den3, E);
        fill_u32_kernel<<<cdiv((size_t)ND * dw, T), blk, 0, stream>>>((unsigned*)agg3, 0u, (size_t)ND * dw);
        edge_agg_kernel<<<cdiv((size_t)E * NHEAD, T), blk, 0, stream>>>(sn_src, sn_dst, ex3, den3, z3, agg3, E, dh);
        elu_add_kernel<<<cdiv((size_t)ND * dw, T), blk, 0, stream>>>(agg3, nullptr, agg3, (size_t)ND * dw);
        ffn_fused_kernel<<<cdiv(ND, 32), blk, 0, stream>>>(agg3, LP(2, 5), LP(2, 6), LP(2, 7), LP(2, 8), y3, ND, dw);
        layernorm_kernel<<<cdiv(ND, 8), blk, 0, stream>>>(y3, LP(2, 9), LP(2, 10), ND, dw);
    }

    // ================= head: news @ wh_w + wh_b =================
    final_head_kernel<<<cdiv(NN, T), blk, 0, stream>>>(y3, wh_w, wh_b, (float*)d_out);
}
